// Mamba_23837068493121
// MI455X (gfx1250) — matrix operand order repaired and checked by reading, not compiled, not run on hardware
//
#include <hip/hip_runtime.h>
#include <hip/hip_bf16.h>
#include <math.h>

// ---------------- model constants ----------------
#define B_SZ     2
#define L_SZ     4096
#define DMODEL   1024
#define DINNER   2048
#define DSTATE   128
#define NHEADS   32
#define HEADDIM  64
#define DCONV    4
#define CHUNK    64
#define NCHUNK   (L_SZ / CHUNK)          // 64
#define CONVDIM  (DINNER + 2 * DSTATE)   // 2304
#define ZCOLS    (2*DINNER + 2*DSTATE + NHEADS) // 4384
#define NTOK     (B_SZ * L_SZ)           // 8192

// ---------------- WMMA / TDM types ----------------
typedef __attribute__((ext_vector_type(16))) __bf16          v16bf;
typedef __attribute__((ext_vector_type(8)))  float           v8f;
typedef __attribute__((ext_vector_type(8)))  unsigned short  ushort8;
typedef __attribute__((ext_vector_type(4)))  unsigned int    v4u;
typedef __attribute__((ext_vector_type(8)))  int             v8i;
typedef __attribute__((ext_vector_type(4)))  int             v4i;

union FragAB {
  v16bf v;
  ushort8 h[2];
  unsigned short u[16];
};

__device__ __forceinline__ unsigned short f2bf(float f) {
  unsigned int x = __float_as_uint(f);
  unsigned int r = x + 0x7FFFu + ((x >> 16) & 1u);   // RNE
  return (unsigned short)(r >> 16);
}

__device__ __forceinline__ v8f wmma_bf16(v16bf a, v16bf b, v8f c) {
  return __builtin_amdgcn_wmma_f32_16x16x32_bf16(
      /*neg_a=*/false, a, /*neg_b=*/false, b,
      /*c_mod=*/(short)0, c, /*reuse_a=*/false, /*reuse_b=*/false);
}

__device__ __forceinline__ float silu_f(float x) {
  return x / (1.f + __expf(-x));
}

// generic pointer -> 32-bit LDS byte offset (ISA: LDS addr = low 32 bits)
__device__ __forceinline__ unsigned lds_off32(const void* p) {
  return (unsigned)(unsigned long long)p;
}

// Async copy of 16B/lane global -> LDS (gfx1250, ASYNCcnt) — fallback path
#define ASYNC_G2L_B128(ldsaddr, goff, base64)                               \
  asm volatile("global_load_async_to_lds_b128 %0, %1, %2"                   \
               :: "v"(ldsaddr), "v"(goff), "s"(base64) : "memory")

#if __has_builtin(__builtin_amdgcn_tensor_load_to_lds) && \
    __has_builtin(__builtin_amdgcn_s_wait_tensorcnt)
#define HAVE_TDM 1
#endif

#ifdef HAVE_TDM
// Issue one TDM 2D tile load: global (row-pitch = stride0 elements, 2B elems)
// -> LDS packed tile of tile1 rows x tile0 elements. D# per ISA ch.8.
// This toolchain's builtin is the 6-arg form:
//   (uint32x4 g0, int32x8 g1, int32x4 g2, int32x4 g3, int32x8 g4, i32 cpol)
__device__ __forceinline__ void tdm_load_2d(
    unsigned lds_addr, unsigned long long gaddr,
    unsigned tdim0, unsigned tdim1,        // tensor extents for OOB zero-fill
    unsigned tile0, unsigned tile1,        // tile dims (elements, rows)
    unsigned long long stride0) {          // row pitch in elements
  v4u g0;
  g0[0] = 1u;                                   // count=1 (valid user D#)
  g0[1] = lds_addr;                             // lds_addr [63:32]
  g0[2] = (unsigned)gaddr;                      // global_addr [95:64]
  g0[3] = (unsigned)(gaddr >> 32) | (2u << 30); // global_addr hi + type=2
  v8i g1;
  g1[0] = 0x10000;                              // data_size=1 (2 bytes)
  g1[1] = (int)((tdim0 & 0xFFFFu) << 16);       // tensor_dim0 lo16
  g1[2] = (int)((tdim0 >> 16) | ((tdim1 & 0xFFFFu) << 16));
  g1[3] = (int)((tdim1 >> 16) | (tile0 << 16)); // tensor_dim1 hi / tile_dim0
  g1[4] = (int)(tile1 & 0xFFFFu);               // tile_dim1; tile_dim2=0
  g1[5] = (int)(unsigned)(stride0 & 0xFFFFFFFFu);
  g1[6] = (int)((unsigned)(stride0 >> 32) & 0xFFFFu); // dim1_stride unused
  g1[7] = 0;
  v4i g2 = {0, 0, 0, 0};
  v4i g3 = {0, 0, 0, 0};
  v8i g4 = {0, 0, 0, 0, 0, 0, 0, 0};
  __builtin_amdgcn_tensor_load_to_lds(g0, g1, g2, g3, g4, 0);
}
#endif

// ---------------- K0: f32 -> bf16 convert ----------------
__global__ __launch_bounds__(256) void f32_to_bf16_kernel(
    const float* __restrict__ in, unsigned short* __restrict__ out, size_t n) {
  size_t i = (size_t)blockIdx.x * blockDim.x + threadIdx.x;
  size_t stride = (size_t)gridDim.x * blockDim.x;
  for (; i < n; i += stride) out[i] = f2bf(in[i]);
}

// ---------------- K1: bf16 GEMM  C[M,N] = A[M,K] * B[N,K]^T ----------------
// 256 threads (8 waves). Block tile 128(M) x 64(N) x 64(K-step).
// Double-buffered LDS staging: Tensor Data Mover (TENSORcnt) when available,
// else GLOBAL_LOAD_ASYNC_TO_LDS_B128 (ASYNCcnt). Next-stage DMA overlaps WMMA.
__global__ __launch_bounds__(256) void gemm_bf16_nt_async(
    const unsigned short* __restrict__ A,   // [M x K] row-major (K contiguous)
    const unsigned short* __restrict__ Bm,  // [N x K] row-major (K contiguous)
    float* __restrict__ C, int M, int N, int K) {
  __shared__ __attribute__((aligned(16))) unsigned short sA[2][128][64];
  __shared__ __attribute__((aligned(16))) unsigned short sB[2][64][64];

  const int tid  = threadIdx.x;
  const int lane = tid & 31;
  const int wave = tid >> 5;
  const int wm = wave >> 1;              // 0..3 (M)
  const int wn = wave & 1;               // 0..1 (N)
  const int lr = lane & 15;
  const int hi = lane >> 4;
  const int mblk = blockIdx.x * 128;
  const int nblk = blockIdx.y * 64;

  const unsigned long long abase = (unsigned long long)A;
  const unsigned long long bbase = (unsigned long long)Bm;

#ifdef HAVE_TDM
  const unsigned brows = (unsigned)((N - nblk) < 64 ? (N - nblk) : 64);
  auto issue_stage = [&](int buf, int k0) {   // wave 0 only: 2 TDM descriptors
    unsigned long long ga = abase + ((unsigned long long)(unsigned)mblk * (unsigned)K +
                                     (unsigned)k0) * 2ull;
    tdm_load_2d(lds_off32(&sA[buf][0][0]), ga, 64u, 128u, 64u, 128u,
                (unsigned long long)(unsigned)K);
    unsigned long long gb = bbase + ((unsigned long long)(unsigned)nblk * (unsigned)K +
                                     (unsigned)k0) * 2ull;
    tdm_load_2d(lds_off32(&sB[buf][0][0]), gb, 64u, brows, 64u, 64u,
                (unsigned long long)(unsigned)K);   // OOB rows zero-filled
  };
#else
  auto issue_stage = [&](int buf, int k0) {   // all waves: 6 async b128 each
    #pragma unroll
    for (int j = 0; j < 4; ++j) {
      int idx = tid + j * 256;
      int r = idx >> 3, c = (idx & 7) * 8;
      unsigned lds = lds_off32(&sA[buf][r][c]);
      unsigned off = ((unsigned)(mblk + r) * (unsigned)K + (unsigned)(k0 + c)) * 2u;
      ASYNC_G2L_B128(lds, off, abase);
    }
    #pragma unroll
    for (int j = 0; j < 2; ++j) {
      int idx = tid + j * 256;
      int r = idx >> 3, c = (idx & 7) * 8;
      int rr = nblk + r; if (rr >= N) rr = N - 1;
      unsigned lds = lds_off32(&sB[buf][r][c]);
      unsigned off = ((unsigned)rr * (unsigned)K + (unsigned)(k0 + c)) * 2u;
      ASYNC_G2L_B128(lds, off, bbase);
    }
  };
#endif

  v8f acc00 = {}, acc01 = {}, acc10 = {}, acc11 = {};
  const int nk = K >> 6;                 // K / 64
#ifdef HAVE_TDM
  if (wave == 0) issue_stage(0, 0);
#else
  issue_stage(0, 0);
#endif
  for (int ks = 0; ks < nk; ++ks) {
    const int buf = ks & 1;
#ifdef HAVE_TDM
    if (wave == 0) {
      if (ks + 1 < nk) {
        issue_stage(buf ^ 1, (ks + 1) << 6);
        __builtin_amdgcn_s_wait_tensorcnt(2);   // stage `buf` landed, next in flight
      } else {
        __builtin_amdgcn_s_wait_tensorcnt(0);
      }
    }
#else
    if (ks + 1 < nk) {
      issue_stage(buf ^ 1, (ks + 1) << 6);
      asm volatile("s_wait_asynccnt 0x6" ::: "memory");
    } else {
      asm volatile("s_wait_asynccnt 0x0" ::: "memory");
    }
#endif
    __syncthreads();
    #pragma unroll
    for (int kk = 0; kk < 64; kk += 32) {
      FragAB a0, a1, b0, b1;
      const unsigned short* pa = &sA[buf][wm * 32 + lr][kk + hi * 8];
      a0.h[0] = *(const ushort8*)pa;        a0.h[1] = *(const ushort8*)(pa + 16);
      pa += 16 * 64;
      a1.h[0] = *(const ushort8*)pa;        a1.h[1] = *(const ushort8*)(pa + 16);
      const unsigned short* pb = &sB[buf][wn * 32 + lr][kk + hi * 8];
      b0.h[0] = *(const ushort8*)pb;        b0.h[1] = *(const ushort8*)(pb + 16);
      pb += 16 * 64;
      b1.h[0] = *(const ushort8*)pb;        b1.h[1] = *(const ushort8*)(pb + 16);
      acc00 = wmma_bf16(a0.v, b0.v, acc00);
      acc01 = wmma_bf16(a0.v, b1.v, acc01);
      acc10 = wmma_bf16(a1.v, b0.v, acc10);
      acc11 = wmma_bf16(a1.v, b1.v, acc11);
    }
    __syncthreads();                       // buffer reusable for next issue
  }
  const int m0 = mblk + wm * 32;
  const int n0 = nblk + wn * 32;
  #pragma unroll
  for (int i = 0; i < 8; ++i) {
    int r0 = m0 + hi * 8 + i;
    int r1 = r0 + 16;
    if (n0 + lr < N) {
      C[(size_t)r0 * N + n0 + lr] = acc00[i];
      C[(size_t)r1 * N + n0 + lr] = acc10[i];
    }
    if (n0 + 16 + lr < N) {
      C[(size_t)r0 * N + n0 + 16 + lr] = acc01[i];
      C[(size_t)r1 * N + n0 + 16 + lr] = acc11[i];
    }
  }
}

// ---------------- K2: causal depthwise conv + SiLU + dt softplus ----------------
__global__ __launch_bounds__(256) void conv_split_kernel(
    const float* __restrict__ zx, const float* __restrict__ conv_w,
    const float* __restrict__ conv_b, const float* __restrict__ dt_bias,
    const float* __restrict__ A_log,
    float* __restrict__ x, float* __restrict__ Bc, float* __restrict__ Cc,
    float* __restrict__ dt, float* __restrict__ dtA) {
  const int row = blockIdx.x;            // b*L + l
  const int l = row & (L_SZ - 1);
  for (int cc = threadIdx.x; cc < CONVDIM; cc += 256) {
    float acc = conv_b[cc];
    #pragma unroll
    for (int j = 0; j < DCONV; ++j) {
      int ls = l - (DCONV - 1) + j;
      if (ls >= 0)
        acc += zx[((size_t)(row - l + ls)) * ZCOLS + DINNER + cc] * conv_w[cc * DCONV + j];
    }
    float v = silu_f(acc);
    if (cc < DINNER)               x[(size_t)row * DINNER + cc] = v;
    else if (cc < DINNER + DSTATE) Bc[(size_t)row * DSTATE + (cc - DINNER)] = v;
    else                           Cc[(size_t)row * DSTATE + (cc - DINNER - DSTATE)] = v;
  }
  if (threadIdx.x < NHEADS) {
    int h = threadIdx.x;
    float raw = zx[(size_t)row * ZCOLS + DINNER + CONVDIM + h] + dt_bias[h];
    float sp = (raw > 20.f) ? raw : log1pf(__expf(raw));
    dt[(size_t)row * NHEADS + h] = sp;
    dtA[(size_t)row * NHEADS + h] = -__expf(A_log[h]) * sp;
  }
}

// ---------------- K3: within-chunk inclusive cumsum of A*dt ----------------
__global__ __launch_bounds__(64) void acum_kernel(
    const float* __restrict__ dtA, float* __restrict__ Acum) {
  const int blk = blockIdx.x;            // ((b*NHEADS + h)*NCHUNK + c)
  const int c = blk & (NCHUNK - 1);
  const int bh = blk >> 6;
  const int h = bh & (NHEADS - 1);
  const int b = bh >> 5;
  __shared__ float s[CHUNK];
  const int t = threadIdx.x;
  s[t] = dtA[((size_t)(b * L_SZ + c * CHUNK + t)) * NHEADS + h];
  __syncthreads();
  for (int off = 1; off < CHUNK; off <<= 1) {
    float v = (t >= off) ? s[t - off] : 0.f;
    __syncthreads();
    s[t] += v;
    __syncthreads();
  }
  Acum[(size_t)blk * CHUNK + t] = s[t];
}

// ---------------- K4: SSD chunk kernel (Y_diag + chunk states) ----------------
__global__ __launch_bounds__(256) void ssd_chunk_kernel(
    const float* __restrict__ x,    // [B,L,H,P]
    const float* __restrict__ dt,   // [B,L,H]
    const float* __restrict__ Bc,   // [B,L,N]
    const float* __restrict__ Cc,   // [B,L,N]
    const float* __restrict__ Acum, // [B,H,NCHUNK,CHUNK]
    float* __restrict__ states,     // [B,NCHUNK,H,P,N]
    float* __restrict__ ypre) {     // [B,L,H*P]
  __shared__ __attribute__((aligned(16))) unsigned short sXT [HEADDIM][CHUNK];
  __shared__ __attribute__((aligned(16))) unsigned short sXTd[HEADDIM][CHUNK];
  __shared__ __attribute__((aligned(16))) unsigned short sB  [CHUNK][DSTATE];
  __shared__ __attribute__((aligned(16))) unsigned short sC  [CHUNK][DSTATE];
  __shared__ __attribute__((aligned(16))) unsigned short sG  [CHUNK][CHUNK];
  __shared__ float sAc[CHUNK];

  const int blk = blockIdx.x;            // ((b*NCHUNK + c)*NHEADS + h)
  const int h = blk & (NHEADS - 1);
  const int c = (blk >> 5) & (NCHUNK - 1);
  const int b = blk >> 11;
  const int tid = threadIdx.x;

  if (tid < CHUNK)
    sAc[tid] = Acum[((size_t)(b * NHEADS + h) * NCHUNK + c) * CHUNK + tid];
  __syncthreads();
  const float Alast = sAc[CHUNK - 1];

  const size_t xbase = ((size_t)(b * L_SZ + c * CHUNK) * NHEADS + h) * HEADDIM;
  for (int idx = tid; idx < CHUNK * HEADDIM; idx += 256) {
    int s = idx >> 6, p = idx & 63;
    float dtv = dt[((size_t)(b * L_SZ + c * CHUNK + s)) * NHEADS + h];
    float v = x[xbase + (size_t)s * NHEADS * HEADDIM + p] * dtv;
    float dec = __expf(Alast - sAc[s]);
    sXT[p][s]  = f2bf(v);
    sXTd[p][s] = f2bf(v * dec);
  }
  for (int idx = tid; idx < CHUNK * DSTATE; idx += 256) {
    int s = idx >> 7, n = idx & 127;
    size_t r = (size_t)(b * L_SZ + c * CHUNK + s) * DSTATE + n;
    sB[s][n] = f2bf(Bc[r]);
    sC[s][n] = f2bf(Cc[r]);
  }
  __syncthreads();

  const int lane = tid & 31, wave = tid >> 5;
  const int lr = lane & 15, hi = lane >> 4;
  const int wm = wave >> 1, wn = wave & 1;

  // ---- G = C @ B^T  (M=64 l, N=64 s, K=128 n), wave tile 16x32
  {
    const int l0 = wm * 16, s0 = wn * 32;
    v8f g0 = {}, g1 = {};
    for (int k0 = 0; k0 < DSTATE; k0 += 32) {
      FragAB a, b0, b1;
      const unsigned short* pa = &sC[l0 + lr][k0 + hi * 8];
      a.h[0] = *(const ushort8*)pa;  a.h[1] = *(const ushort8*)(pa + 16);
      const unsigned short* pb = &sB[s0 + lr][k0 + hi * 8];
      b0.h[0] = *(const ushort8*)pb; b0.h[1] = *(const ushort8*)(pb + 16);
      pb = &sB[s0 + 16 + lr][k0 + hi * 8];
      b1.h[0] = *(const ushort8*)pb; b1.h[1] = *(const ushort8*)(pb + 16);
      g0 = wmma_bf16(a.v, b0.v, g0);
      g1 = wmma_bf16(a.v, b1.v, g1);
    }
    #pragma unroll
    for (int i = 0; i < 8; ++i) {
      int l = l0 + hi * 8 + i;
      int s1 = s0 + lr;
      sG[l][s1] = f2bf((s1 <= l) ? g0[i] * __expf(sAc[l] - sAc[s1]) : 0.f);
      int s2 = s0 + 16 + lr;
      sG[l][s2] = f2bf((s2 <= l) ? g1[i] * __expf(sAc[l] - sAc[s2]) : 0.f);
    }
  }
  __syncthreads();

  // ---- Y_diag = Gmask @ Xdt  (M=64 l, N=64 p, K=64 s)
  {
    const int l0 = wm * 16, p0 = wn * 32;
    v8f y0 = {}, y1 = {};
    for (int k0 = 0; k0 < CHUNK; k0 += 32) {
      FragAB a, b0, b1;
      const unsigned short* pa = &sG[l0 + lr][k0 + hi * 8];
      a.h[0] = *(const ushort8*)pa;  a.h[1] = *(const ushort8*)(pa + 16);
      const unsigned short* pb = &sXT[p0 + lr][k0 + hi * 8];
      b0.h[0] = *(const ushort8*)pb; b0.h[1] = *(const ushort8*)(pb + 16);
      pb = &sXT[p0 + 16 + lr][k0 + hi * 8];
      b1.h[0] = *(const ushort8*)pb; b1.h[1] = *(const ushort8*)(pb + 16);
      y0 = wmma_bf16(a.v, b0.v, y0);
      y1 = wmma_bf16(a.v, b1.v, y1);
    }
    #pragma unroll
    for (int i = 0; i < 8; ++i) {
      int l = l0 + hi * 8 + i;
      size_t row = (size_t)(b * L_SZ + c * CHUNK + l) * DINNER + h * HEADDIM;
      ypre[row + p0 + lr]      = y0[i];
      ypre[row + p0 + 16 + lr] = y1[i];
    }
  }

  // ---- states = (Xdt^T diag(decay)) @ B  (M=64 p, N=128 n, K=64 s)
  {
    const int p0 = wm * 16, nb = wn * 64;
    v8f acc[4] = {{}, {}, {}, {}};
    for (int k0 = 0; k0 < CHUNK; k0 += 32) {
      FragAB a;
      const unsigned short* pa = &sXTd[p0 + lr][k0 + hi * 8];
      a.h[0] = *(const ushort8*)pa;  a.h[1] = *(const ushort8*)(pa + 16);
      #pragma unroll
      for (int t = 0; t < 4; ++t) {
        FragAB bt;
        #pragma unroll
        for (int i = 0; i < 16; ++i)
          bt.u[i] = sB[k0 + hi * 8 + (i & 7) + (i >> 3) * 16][nb + t * 16 + lr];
        acc[t] = wmma_bf16(a.v, bt.v, acc[t]);
      }
    }
    size_t base = ((size_t)(b * NCHUNK + c) * NHEADS + h) * (HEADDIM * DSTATE);
    #pragma unroll
    for (int t = 0; t < 4; ++t) {
      int n = nb + t * 16 + lr;
      #pragma unroll
      for (int i = 0; i < 8; ++i) {
        int p = p0 + hi * 8 + i;
        states[base + (size_t)p * DSTATE + n] = acc[t][i];
      }
    }
  }
}

// ---------------- K5: sequential inter-chunk state recurrence ----------------
__global__ __launch_bounds__(256) void state_recur_kernel(
    const float* __restrict__ Acum, float* __restrict__ states) {
  const int bh = blockIdx.x;             // b*NHEADS + h
  const int h = bh & (NHEADS - 1);
  const int b = bh >> 5;
  const int tid = threadIdx.x;
  float S[32];
  #pragma unroll
  for (int k = 0; k < 32; ++k) S[k] = 0.f;
  for (int c = 0; c < NCHUNK; ++c) {
    float dec = __expf(Acum[((size_t)(b * NHEADS + h) * NCHUNK + c) * CHUNK + (CHUNK - 1)]);
    size_t base = ((size_t)(b * NCHUNK + c) * NHEADS + h) * (HEADDIM * DSTATE);
    #pragma unroll
    for (int k = 0; k < 32; ++k) {
      size_t off = base + tid + k * 256;
      float tmp = states[off];
      states[off] = S[k];                // prev state for this chunk
      S[k] = S[k] * dec + tmp;
    }
  }
}

// ---------------- K6: Y_off + D skip, accumulate into ypre ----------------
__global__ __launch_bounds__(256) void ssd_yoff_kernel(
    const float* __restrict__ Cc, const float* __restrict__ Acum,
    const float* __restrict__ prev,     // [B,NCHUNK,H,P,N] (states_prev)
    const float* __restrict__ x, const float* __restrict__ Dp,
    float* __restrict__ ypre) {
  __shared__ __attribute__((aligned(16))) unsigned short sCs[CHUNK][DSTATE];
  __shared__ __attribute__((aligned(16))) unsigned short sP [HEADDIM][DSTATE];
  __shared__ float sAc[CHUNK];

  const int blk = blockIdx.x;
  const int h = blk & (NHEADS - 1);
  const int c = (blk >> 5) & (NCHUNK - 1);
  const int b = blk >> 11;
  const int tid = threadIdx.x;

  if (tid < CHUNK)
    sAc[tid] = Acum[((size_t)(b * NHEADS + h) * NCHUNK + c) * CHUNK + tid];
  __syncthreads();

  for (int idx = tid; idx < CHUNK * DSTATE; idx += 256) {
    int s = idx >> 7, n = idx & 127;
    sCs[s][n] = f2bf(Cc[(size_t)(b * L_SZ + c * CHUNK + s) * DSTATE + n] * __expf(sAc[s]));
  }
  size_t pbase = ((size_t)(b * NCHUNK + c) * NHEADS + h) * (HEADDIM * DSTATE);
  for (int idx = tid; idx < HEADDIM * DSTATE; idx += 256) {
    int p = idx >> 7, n = idx & 127;
    sP[p][n] = f2bf(prev[pbase + (size_t)p * DSTATE + n]);
  }
  __syncthreads();

  const int lane = tid & 31, wave = tid >> 5;
  const int lr = lane & 15, hi = lane >> 4;
  const int wm = wave >> 1, wn = wave & 1;
  const int l0 = wm * 16, p0 = wn * 32;

  v8f y0 = {}, y1 = {};
  for (int k0 = 0; k0 < DSTATE; k0 += 32) {
    FragAB a, b0, b1;
    const unsigned short* pa = &sCs[l0 + lr][k0 + hi * 8];
    a.h[0] = *(const ushort8*)pa;  a.h[1] = *(const ushort8*)(pa + 16);
    const unsigned short* pb = &sP[p0 + lr][k0 + hi * 8];
    b0.h[0] = *(const ushort8*)pb; b0.h[1] = *(const ushort8*)(pb + 16);
    pb = &sP[p0 + 16 + lr][k0 + hi * 8];
    b1.h[0] = *(const ushort8*)pb; b1.h[1] = *(const ushort8*)(pb + 16);
    y0 = wmma_bf16(a.v, b0.v, y0);
    y1 = wmma_bf16(a.v, b1.v, y1);
  }
  const float Dh = Dp[h];
  #pragma unroll
  for (int i = 0; i < 8; ++i) {
    int l = l0 + hi * 8 + i;
    size_t row = (size_t)(b * L_SZ + c * CHUNK + l);
    size_t xrow = (row * NHEADS + h) * HEADDIM;
    size_t yrow = row * DINNER + h * HEADDIM;
    {
      int p = p0 + lr;
      ypre[yrow + p] += y0[i] + x[xrow + p] * Dh;
    }
    {
      int p = p0 + 16 + lr;
      ypre[yrow + p] += y1[i] + x[xrow + p] * Dh;
    }
  }
}

// ---------------- K7: gating (silu(z)) + RMSNorm -> bf16 ----------------
__global__ __launch_bounds__(256) void gate_norm_kernel(
    const float* __restrict__ zx, const float* __restrict__ ypre,
    const float* __restrict__ norm_w, unsigned short* __restrict__ ybf) {
  const int row = blockIdx.x;            // b*L + l
  __shared__ float red[8];
  float vals[8];
  float ss = 0.f;
  #pragma unroll
  for (int k = 0; k < 8; ++k) {
    int i = threadIdx.x + k * 256;
    float z = zx[(size_t)row * ZCOLS + i];
    float v = ypre[(size_t)row * DINNER + i] * silu_f(z);
    vals[k] = v;
    ss += v * v;
  }
  #pragma unroll
  for (int off = 16; off > 0; off >>= 1) ss += __shfl_xor(ss, off, 32);
  const int lane = threadIdx.x & 31, wave = threadIdx.x >> 5;
  if (lane == 0) red[wave] = ss;
  __syncthreads();
  if (threadIdx.x == 0) {
    float t = 0.f;
    #pragma unroll
    for (int j = 0; j < 8; ++j) t += red[j];
    red[0] = t;
  }
  __syncthreads();
  float scale = rsqrtf(red[0] / (float)DINNER + 1e-5f);
  #pragma unroll
  for (int k = 0; k < 8; ++k) {
    int i = threadIdx.x + k * 256;
    ybf[(size_t)row * DINNER + i] = f2bf(vals[k] * scale * norm_w[i]);
  }
}

// ---------------- host-side launcher ----------------
extern "C" void kernel_launch(void* const* d_in, const int* in_sizes, int n_in,
                              void* d_out, int out_size, void* d_ws, size_t ws_size,
                              hipStream_t stream) {
  (void)in_sizes; (void)n_in; (void)out_size; (void)ws_size;
  const float* u       = (const float*)d_in[0];
  const float* W_in    = (const float*)d_in[1];
  const float* conv_w  = (const float*)d_in[2];
  const float* conv_b  = (const float*)d_in[3];
  const float* dt_bias = (const float*)d_in[4];
  const float* A_log   = (const float*)d_in[5];
  const float* D_par   = (const float*)d_in[6];
  const float* norm_w  = (const float*)d_in[7];
  const float* W_out   = (const float*)d_in[8];
  float* out = (float*)d_out;

  char* ws = (char*)d_ws;
  size_t off = 0;
  auto alloc = [&](size_t bytes) -> char* {
    char* p = ws + off;
    off = (off + bytes + 255) & ~(size_t)255;
    return p;
  };
  unsigned short* ub    = (unsigned short*)alloc((size_t)NTOK * DMODEL * 2);
  unsigned short* winb  = (unsigned short*)alloc((size_t)ZCOLS * DMODEL * 2);
  unsigned short* woutb = (unsigned short*)alloc((size_t)DMODEL * DINNER * 2);
  float* zx   = (float*)alloc((size_t)NTOK * ZCOLS * 4);
  float* xbuf = (float*)alloc((size_t)NTOK * DINNER * 4);
  float* Bcb  = (float*)alloc((size_t)NTOK * DSTATE * 4);
  float* Ccb  = (float*)alloc((size_t)NTOK * DSTATE * 4);
  float* dtb  = (float*)alloc((size_t)NTOK * NHEADS * 4);
  float* dtA  = (float*)alloc((size_t)NTOK * NHEADS * 4);
  float* Acum = (float*)alloc((size_t)B_SZ * NHEADS * NCHUNK * CHUNK * 4);
  float* st   = (float*)alloc((size_t)B_SZ * NCHUNK * NHEADS * HEADDIM * DSTATE * 4);
  float* ypre = (float*)alloc((size_t)NTOK * DINNER * 4);
  unsigned short* ybf = (unsigned short*)alloc((size_t)NTOK * DINNER * 2);

  // 0) converts
  f32_to_bf16_kernel<<<2048, 256, 0, stream>>>(u, ub, (size_t)NTOK * DMODEL);
  f32_to_bf16_kernel<<<2048, 256, 0, stream>>>(W_in, winb, (size_t)ZCOLS * DMODEL);
  f32_to_bf16_kernel<<<512, 256, 0, stream>>>(W_out, woutb, (size_t)DMODEL * DINNER);

  // 1) zxbcdt = u @ W_in^T
  {
    dim3 g(NTOK / 128, (ZCOLS + 63) / 64);
    gemm_bf16_nt_async<<<g, 256, 0, stream>>>(ub, winb, zx, NTOK, ZCOLS, DMODEL);
  }
  // 2) conv + silu + splits + softplus(dt)
  conv_split_kernel<<<NTOK, 256, 0, stream>>>(zx, conv_w, conv_b, dt_bias, A_log,
                                              xbuf, Bcb, Ccb, dtb, dtA);
  // 3) per-chunk cumsum of A*dt
  acum_kernel<<<B_SZ * NHEADS * NCHUNK, 64, 0, stream>>>(dtA, Acum);
  // 4) Y_diag + chunk states (WMMA)
  ssd_chunk_kernel<<<B_SZ * NCHUNK * NHEADS, 256, 0, stream>>>(
      xbuf, dtb, Bcb, Ccb, Acum, st, ypre);
  // 5) inter-chunk recurrence -> states_prev (in place)
  state_recur_kernel<<<B_SZ * NHEADS, 256, 0, stream>>>(Acum, st);
  // 6) Y_off + D skip (WMMA)
  ssd_yoff_kernel<<<B_SZ * NCHUNK * NHEADS, 256, 0, stream>>>(
      Ccb, Acum, st, xbuf, D_par, ypre);
  // 7) gated RMSNorm -> bf16
  gate_norm_kernel<<<NTOK, 256, 0, stream>>>(zx, ypre, norm_w, ybf);
  // 8) out = y @ W_out^T
  {
    dim3 g(NTOK / 128, DMODEL / 64);
    gemm_bf16_nt_async<<<g, 256, 0, stream>>>(ybf, woutb, out, NTOK, DMODEL, DINNER);
  }
}
